// SCN_33749853012578
// MI455X (gfx1250) — compile-verified, matrix operand order checked by reading, not run on hardware
//
#include <hip/hip_runtime.h>

// ---------------------------------------------------------------------------
// CDNA5 (gfx1250) WMMA fragments: bf16 A/B, f32 accumulate, 16x16x32 tiles.
// ---------------------------------------------------------------------------
typedef __attribute__((ext_vector_type(16))) __bf16 v16bf;
typedef __attribute__((ext_vector_type(8)))  float  v8f;
typedef __attribute__((ext_vector_type(4)))  unsigned int v4u;
typedef __attribute__((ext_vector_type(8)))  int    v8i;
typedef __attribute__((ext_vector_type(4)))  int    v4i;

#if __has_builtin(__builtin_amdgcn_tensor_load_to_lds)
#define HAVE_TDM 1
#else
#define HAVE_TDM 0
#endif

union FragU { v16bf v; uint4 q[2]; };

__device__ __forceinline__ v16bf frag_zero() {
  FragU u;
  u.q[0] = make_uint4(0u, 0u, 0u, 0u);
  u.q[1] = make_uint4(0u, 0u, 0u, 0u);
  return u.v;
}

// A fragment: lane owns row (lane&15). Elements j0..7 at p[0..7] (K 0..7 (+8 for
// hi lanes)), j8..15 at p[16..23] (K 16..23 (+8)). Two 16B loads.
__device__ __forceinline__ v16bf load_a_frag(const __bf16* p) {
  FragU u;
  u.q[0] = *(const uint4*)(p);
  u.q[1] = *(const uint4*)(p + 16);
  return u.v;
}

// B fragment: packed layout, 16 contiguous bf16 per lane. Two 16B loads.
__device__ __forceinline__ v16bf load_b_frag(const __bf16* p) {
  FragU u;
  u.q[0] = *(const uint4*)(p);
  u.q[1] = *(const uint4*)(p + 8);
  return u.v;
}

#if HAVE_TDM
// Issue a 1-D contiguous TDM copy: n8 elements of 8 bytes, global -> LDS.
// D# group0: count=1, lds_addr, 57b global addr, type=2 ("image").
// D# group1: data_size=3 (8B), tensor_dim0=n8, tensor_dim1=1, tile_dim0=n8,
//            tile_dim1=1, tensor_dim0_stride=n8.
__device__ __forceinline__ void tdm_load_1d(const __bf16* gptr, unsigned lds_addr,
                                            unsigned n8) {
  unsigned long long ga = (unsigned long long)(size_t)gptr;
  v4u g0;
  g0[0] = 1u;                                              // count=1
  g0[1] = lds_addr;
  g0[2] = (unsigned)(ga & 0xffffffffu);
  g0[3] = (unsigned)((ga >> 32) & 0x1ffffffu) | (2u << 30); // type=2
  v8i g1;
  g1[0] = (int)(3u << 16);                                 // data_size = 8B
  g1[1] = (int)((n8 & 0xffffu) << 16);                     // tensor_dim0[15:0]
  g1[2] = (int)((n8 >> 16) | (1u << 16));                  // dim0[31:16], dim1=1
  g1[3] = (int)((n8 & 0xffffu) << 16);                     // tile_dim0 = n8
  g1[4] = 1;                                               // tile_dim1 = 1
  g1[5] = (int)n8;                                         // dim0_stride = n8
  g1[6] = 0;
  g1[7] = 0;
  v4i z4 = {0, 0, 0, 0};
#if __clang_major__ >= 23
  v8i z8 = {0, 0, 0, 0, 0, 0, 0, 0};
  __builtin_amdgcn_tensor_load_to_lds(g0, g1, z4, z4, z8, 0);
#else
  __builtin_amdgcn_tensor_load_to_lds(g0, g1, z4, z4, 0);
#endif
}
#endif

// ---------------------------------------------------------------------------
// Kernels
// ---------------------------------------------------------------------------

// 27-offset submanifold sparse conv, rulebook gather -> WMMA.
// xp : bf16 [(N+1) x Cpad], row N zeros (pad row for missing neighbors)
// Bp : packed weights, nt-major: [Nt][27][Kt][32 lanes x 16 bf16]
// Per (nt,off) the Kt B-tiles are contiguous (Kt*1024 bytes) and are staged
// into LDS by the Tensor Data Mover, double-buffered against compute.
__global__ void k_subconv(const __bf16* __restrict__ xp, const int* __restrict__ nbr,
                          const __bf16* __restrict__ Bp, float* __restrict__ out,
                          int N, int Cpad, int Cout, int Nt) {
  __shared__ __align__(16) __bf16 smB[2][14 * 512];   // 2 x 14KB (Kt <= 14)
  const int wid  = threadIdx.x >> 5;
  const int lane = threadIdx.x & 31;
  const int tm   = blockIdx.x * 8 + wid;              // 8 waves: rows tm*16..+15
  const int nt   = blockIdx.y;
  const int lhi  = lane >> 4;
  const int m    = lane & 15;
  const int row  = tm * 16 + m;
  const int Kt   = Cpad >> 5;
  const int chunk = Kt << 9;                          // bf16 elems per (nt,off)
  const __bf16* bbase = Bp + (size_t)nt * 27 * chunk;

#if HAVE_TDM
  if (threadIdx.x < 32) {
    tdm_load_1d(bbase, (unsigned)(size_t)(void*)smB[0], (unsigned)(chunk >> 2));
    __builtin_amdgcn_s_wait_tensorcnt(0);
  }
  __syncthreads();
#else
  for (int i = threadIdx.x; i < (chunk >> 3); i += (int)blockDim.x)
    ((uint4*)smB[0])[i] = ((const uint4*)bbase)[i];
  __syncthreads();
#endif

  v8f acc = {};
  for (int off = 0; off < 27; ++off) {
    const int cur = off & 1;
#if HAVE_TDM
    if (off + 1 < 27 && threadIdx.x < 32)
      tdm_load_1d(bbase + (size_t)(off + 1) * chunk,
                  (unsigned)(size_t)(void*)smB[cur ^ 1], (unsigned)(chunk >> 2));
#endif
    int g = (row < N) ? nbr[(size_t)row * 27 + off] : N;   // N => zero pad row
    if (off + 1 < 27 && row < N) {                         // warm next gather row
      int g2 = nbr[(size_t)row * 27 + off + 1];
      __builtin_prefetch((const void*)(xp + (size_t)g2 * Cpad), 0, 0);
    }
    const __bf16* ap = xp + (size_t)g * Cpad + lhi * 8;
    const __bf16* bp = smB[cur] + lane * 16;
    for (int kt = 0; kt < Kt; ++kt) {
      v16bf a = load_a_frag(ap + kt * 32);
      v16bf b = load_b_frag(bp + (size_t)kt * 512);
      acc = __builtin_amdgcn_wmma_f32_16x16x32_bf16(false, a, false, b,
                                                    (short)0, acc, false, false);
    }
#if HAVE_TDM
    if (threadIdx.x < 32) __builtin_amdgcn_s_wait_tensorcnt(0);
    __syncthreads();
#else
    if (off + 1 < 27) {
      __syncthreads();
      for (int i = threadIdx.x; i < (chunk >> 3); i += (int)blockDim.x)
        ((uint4*)smB[cur ^ 1])[i] =
            ((const uint4*)(bbase + (size_t)(off + 1) * chunk))[i];
      __syncthreads();
    }
#endif
  }

  const int col = nt * 16 + m;
  if (col < Cout) {
    for (int r = 0; r < 8; ++r) {
      int orow = tm * 16 + r + lhi * 8;
      if (orow < N) out[(size_t)orow * Cout + col] = acc[r];
    }
  }
}

// Generic WMMA GEMM with optional row gather (A side) / scatter (out side),
// optional bias / relu / accumulate.  flags: 1=accumulate, 2=relu.
// B packed layout: [Kt][Nt][32 lanes x 16 bf16] (global, L2-resident).
__global__ void k_gemm(const __bf16* __restrict__ A, int M,
                       const int* __restrict__ gidx,
                       const __bf16* __restrict__ Bp,
                       const float* __restrict__ bias,
                       float* __restrict__ out,
                       const int* __restrict__ sidx,
                       int Kt, int ldA, int Cout, int ldOut, int Nt, int flags) {
  const int wid  = threadIdx.x >> 5;
  const int lane = threadIdx.x & 31;
  const int tm   = blockIdx.x * 4 + wid;
  if (tm * 16 >= M) return;                 // wave-uniform, no barriers here
  const int nt  = blockIdx.y;
  const int lhi = lane >> 4;
  const int m   = lane & 15;
  const int r   = tm * 16 + m;
  const bool valid = (r < M);
  const int arow = valid ? (gidx ? gidx[r] : r) : 0;
  const __bf16* ap = A + (size_t)arow * ldA + lhi * 8;
  const __bf16* bp = Bp + (size_t)nt * 512 + lane * 16;
  v8f acc = {};
  for (int kt = 0; kt < Kt; ++kt) {
    v16bf a = valid ? load_a_frag(ap + kt * 32) : frag_zero();
    v16bf b = load_b_frag(bp + (size_t)kt * Nt * 512);
    acc = __builtin_amdgcn_wmma_f32_16x16x32_bf16(false, a, false, b,
                                                  (short)0, acc, false, false);
  }
  const int col = nt * 16 + m;
  if (col < Cout) {
    const float bv = bias ? bias[col] : 0.0f;
    for (int rr = 0; rr < 8; ++rr) {
      int rowi = tm * 16 + rr + lhi * 8;
      if (rowi < M) {
        int orow = sidx ? sidx[rowi] : rowi;
        float v = acc[rr] + bv;
        if (flags & 2) v = fmaxf(v, 0.0f);
        float* o = out + (size_t)orow * ldOut + col;
        if (flags & 1) *o += v; else *o = v;
      }
    }
  }
}

// Repack [Ksrc x Co] f32 weight slices into WMMA-B lane layout, bf16,
// zero-filled for k>=Ksrc or n>=Co.
// ntMajor=0: [slice][Kt][Nt][512]   (generic GEMM)
// ntMajor=1: [Nt][slice][Kt][512]   (subconv; contiguous per (nt,off) for TDM)
__global__ void k_pack_b(const float* __restrict__ W, __bf16* __restrict__ out,
                         int Kt, int Nt, int Ksrc, int Co, int slices,
                         int ntMajor, int total) {
  int tid = blockIdx.x * blockDim.x + threadIdx.x;
  if (tid >= total) return;
  int lane = tid & 31;
  int t = tid >> 5;
  int nt = t % Nt; t /= Nt;
  int kt = t % Kt;
  int s  = t / Kt;
  int n   = nt * 16 + (lane & 15);
  int khi = (lane >= 16) ? 8 : 0;
  const float* Ws = W + (size_t)s * Ksrc * Co;
  union { __bf16 v[16]; uint4 q[2]; } u;
  for (int j = 0; j < 16; ++j) {
    int k = kt * 32 + j + ((j >= 8) ? 8 : 0) + khi;
    float v = (k < Ksrc && n < Co) ? Ws[(size_t)k * Co + n] : 0.0f;
    u.v[j] = (__bf16)v;
  }
  size_t slot = ntMajor ? ((((size_t)nt * slices + s) * Kt + kt) * 32 + lane)
                        : ((((size_t)s * Kt + kt) * Nt + nt) * 32 + lane);
  uint4* dst = (uint4*)(out + slot * 16);
  dst[0] = u.q[0];
  dst[1] = u.q[1];
}

// Per-channel BN statistics over N rows -> fused scale/shift (LDS reduction).
__global__ void k_bn_stats(const float* __restrict__ x, int N, int C,
                           const float* __restrict__ g, const float* __restrict__ b,
                           float* __restrict__ scale, float* __restrict__ shift) {
  const int c = blockIdx.x;
  __shared__ float s1[256];
  __shared__ float s2[256];
  float sum = 0.0f, sum2 = 0.0f;
  for (int n = threadIdx.x; n < N; n += blockDim.x) {
    float v = x[(size_t)n * C + c];
    sum += v; sum2 += v * v;
  }
  s1[threadIdx.x] = sum; s2[threadIdx.x] = sum2;
  __syncthreads();
  for (int st = 128; st > 0; st >>= 1) {
    if (threadIdx.x < st) {
      s1[threadIdx.x] += s1[threadIdx.x + st];
      s2[threadIdx.x] += s2[threadIdx.x + st];
    }
    __syncthreads();
  }
  if (threadIdx.x == 0) {
    float mu  = s1[0] / (float)N;
    float var = s2[0] / (float)N - mu * mu;
    float sc  = g[c] * rsqrtf(var + 1e-4f);
    scale[c] = sc;
    shift[c] = b[c] - mu * sc;
  }
}

// y[(N+1) x C] bf16 = relu(x*scale+shift); row N zeroed (pad row).
__global__ void k_bn_apply(const float* __restrict__ x, const float* __restrict__ scale,
                           const float* __restrict__ shift, __bf16* __restrict__ y,
                           int N, int C, int total) {
  int i = blockIdx.x * blockDim.x + threadIdx.x;
  if (i >= total) return;
  int row = i / C, col = i - row * C;
  float v = 0.0f;
  if (row < N) v = fmaxf(x[(size_t)row * C + col] * scale[col] + shift[col], 0.0f);
  y[i] = (__bf16)v;
}

// dst[(N+1) x Cd] bf16 = src[N x Cs] zero-extended in cols, pad row zeroed.
__global__ void k_pad_bf16(const float* __restrict__ src, __bf16* __restrict__ dst,
                           int N, int Cs, int Cd, int total) {
  int i = blockIdx.x * blockDim.x + threadIdx.x;
  if (i >= total) return;
  int row = i / Cd, col = i - row * Cd;
  float v = (row < N && col < Cs) ? src[(size_t)row * Cs + col] : 0.0f;
  dst[i] = (__bf16)v;
}

__global__ void k_add(float* __restrict__ dst, const float* __restrict__ src, int n) {
  int i = blockIdx.x * blockDim.x + threadIdx.x;
  if (i < n) dst[i] += src[i];
}

__global__ void k_zero(float* __restrict__ p, int n) {
  int i = blockIdx.x * blockDim.x + threadIdx.x;
  if (i < n) p[i] = 0.0f;
}

// dst[row*ld + col] = src[row*C + col]  (skip-copy into concat buffer)
__global__ void k_copy(const float* __restrict__ src, float* __restrict__ dst,
                       int C, int ld, int total) {
  int i = blockIdx.x * blockDim.x + threadIdx.x;
  if (i >= total) return;
  int row = i / C, col = i - row * C;
  dst[(size_t)row * ld + col] = src[i];
}

// ---------------------------------------------------------------------------
// Host-side orchestration
// ---------------------------------------------------------------------------
struct BnP  { const float* g; const float* b; };
struct BlkP { BnP bn1; const float* w1; BnP bn2; const float* w2; const float* nin; };
struct LvlP {
  BlkP blocks[2];
  BnP bn_down; const float* down;
  BnP bn_up;   const float* up;
  BlkP blocks2[2];
};

static const int PL[7] = {32, 64, 96, 128, 160, 192, 224};

struct Ctx {
  hipStream_t s;
  char* ws; size_t off, cap;
  int Nl[7];
  const int* nbr[7];
  const int* didx[6][8];
  const int* dpar[6][8];
  int dlen[6][8];
  LvlP lv[7];
  const float* stem;
  BnP bn_out;
  const float *lin_w, *lin_b, *lin1_w, *lin1_b, *cen1_w, *cen1_b, *cen2_w, *cen2_b;
  void* alloc(size_t bytes) {
    size_t a = (off + 255) & ~(size_t)255;
    off = a + bytes;
    return (void*)(ws + a);
  }
};

static inline unsigned cdiv(size_t a, size_t b) { return (unsigned)((a + b - 1) / b); }

static void L_zero(Ctx& c, float* p, size_t n) {
  if (n) k_zero<<<cdiv(n, 256), 256, 0, c.s>>>(p, (int)n);
}
static void L_add(Ctx& c, float* d, const float* s, size_t n) {
  if (n) k_add<<<cdiv(n, 256), 256, 0, c.s>>>(d, s, (int)n);
}

static __bf16* pack_w(Ctx& c, const float* W, int slices, int Ksrc, int Kt, int Co,
                      int ntMajor) {
  int Nt = (Co + 15) / 16;
  __bf16* out = (__bf16*)c.alloc((size_t)slices * Kt * Nt * 512 * sizeof(__bf16));
  int total = slices * Kt * Nt * 32;
  k_pack_b<<<cdiv(total, 256), 256, 0, c.s>>>(W, out, Kt, Nt, Ksrc, Co, slices,
                                              ntMajor, total);
  return out;
}

static __bf16* bnrelu_bf(Ctx& c, const float* x, int N, int C, BnP bn) {
  float* scale = (float*)c.alloc((size_t)C * sizeof(float));
  float* shift = (float*)c.alloc((size_t)C * sizeof(float));
  k_bn_stats<<<dim3(C), dim3(256), 0, c.s>>>(x, N, C, bn.g, bn.b, scale, shift);
  __bf16* y = (__bf16*)c.alloc((size_t)(N + 1) * C * sizeof(__bf16));
  int total = (N + 1) * C;
  k_bn_apply<<<cdiv(total, 256), 256, 0, c.s>>>(x, scale, shift, y, N, C, total);
  return y;
}

static __bf16* to_bf16_pad(Ctx& c, const float* src, int N, int Cs, int Cd) {
  __bf16* dst = (__bf16*)c.alloc((size_t)(N + 1) * Cd * sizeof(__bf16));
  int total = (N + 1) * Cd;
  k_pad_bf16<<<cdiv(total, 256), 256, 0, c.s>>>(src, dst, N, Cs, Cd, total);
  return dst;
}

static void subconv(Ctx& c, const __bf16* xp, const int* nbr, const __bf16* Bp,
                    float* out, int N, int Cpad, int Cout) {
  int Nt = (Cout + 15) / 16;
  dim3 g(cdiv((size_t)N, 128), Nt);        // 8 waves/block, 16 rows each
  k_subconv<<<g, dim3(256), 0, c.s>>>(xp, nbr, Bp, out, N, Cpad, Cout, Nt);
}

static void gemm(Ctx& c, const __bf16* A, int M, const int* gidx, const __bf16* Bp,
                 const float* bias, float* out, const int* sidx,
                 int K, int Cout, int ldOut, int flags) {
  if (M <= 0) return;
  int Kt = K / 32, Nt = (Cout + 15) / 16;
  dim3 g(cdiv((size_t)M, 64), Nt);
  k_gemm<<<g, dim3(128), 0, c.s>>>(A, M, gidx, Bp, bias, out, sidx,
                                   Kt, K, Cout, ldOut, Nt, flags);
}

// Pre-activation residual block: out = shortcut(x) + conv2(bnrelu(conv1(bnrelu(x))))
static void run_block(Ctx& c, const BlkP& p, const float* xin, float* xout,
                      int N, int a, int b, const int* nbr) {
  size_t mk = c.off;
  __bf16* t1 = bnrelu_bf(c, xin, N, a, p.bn1);
  __bf16* W1 = pack_w(c, p.w1, 27, a, a / 32, b, /*ntMajor*/1);
  float* h = (float*)c.alloc((size_t)N * b * sizeof(float));
  subconv(c, t1, nbr, W1, h, N, a, b);
  __bf16* t2 = bnrelu_bf(c, h, N, b, p.bn2);
  __bf16* W2 = pack_w(c, p.w2, 27, b, b / 32, b, /*ntMajor*/1);
  subconv(c, t2, nbr, W2, xout, N, b, b);
  if (p.nin) {
    __bf16* xb = to_bf16_pad(c, xin, N, a, a);
    __bf16* Wn = pack_w(c, p.nin, 1, a, a / 32, b, /*ntMajor*/0);
    gemm(c, xb, N, nullptr, Wn, nullptr, xout, nullptr, a, b, b, /*accum*/1);
  } else {
    L_add(c, xout, xin, (size_t)N * b);
  }
  c.off = mk;
}

static float* run_u(Ctx& c, int lvl, float* x) {
  const int N = c.Nl[lvl], p = PL[lvl];
  float* tmp = (float*)c.alloc((size_t)N * p * sizeof(float));
  float* cur = x;
  for (int i = 0; i < 2; ++i) {
    run_block(c, c.lv[lvl].blocks[i], cur, tmp, N, p, p, c.nbr[lvl]);
    float* t = cur; cur = tmp; tmp = t;
  }
  if (lvl >= 6) return cur;

  float* skip = cur;
  const int N1 = c.Nl[lvl + 1], p1 = PL[lvl + 1];
  float* y = (float*)c.alloc((size_t)N * 2 * p * sizeof(float));   // concat buffer
  size_t mY = c.off;

  // --- downconv: h1[par] += bnrelu(x)[idx] @ down[k] ---
  float* h1 = (float*)c.alloc((size_t)N1 * p1 * sizeof(float));
  {
    size_t mk = c.off;
    __bf16* t = bnrelu_bf(c, cur, N, p, c.lv[lvl].bn_down);
    L_zero(c, h1, (size_t)N1 * p1);
    for (int k = 0; k < 8; ++k) {
      __bf16* Wk = pack_w(c, c.lv[lvl].down + (size_t)k * p * p1, 1, p, p / 32, p1, 0);
      gemm(c, t, c.dlen[lvl][k], c.didx[lvl][k], Wk, nullptr,
           h1, c.dpar[lvl][k], p, p1, p1, /*accum*/1);
    }
    c.off = mk;
  }

  float* r = run_u(c, lvl + 1, h1);

  // --- deconv + concat: y = [skip | bnrelu(r)[par] @ up[k] -> rows idx] ---
  {
    size_t mk = c.off;
    __bf16* t = bnrelu_bf(c, r, N1, p1, c.lv[lvl].bn_up);
    int total = N * p;
    k_copy<<<cdiv(total, 256), 256, 0, c.s>>>(skip, y, p, 2 * p, total);
    for (int k = 0; k < 8; ++k) {
      __bf16* Wk = pack_w(c, c.lv[lvl].up + (size_t)k * p1 * p, 1, p1, p1 / 32, p, 0);
      gemm(c, t, c.dlen[lvl][k], c.dpar[lvl][k], Wk, nullptr,
           y + p, c.didx[lvl][k], p1, p, 2 * p, 0);
    }
    c.off = mk;
  }
  c.off = mY;   // release h1 + entire inner sub-tree; y/skip/tmp persist

  float* b1 = (float*)c.alloc((size_t)N * p * sizeof(float));
  float* b2 = (float*)c.alloc((size_t)N * p * sizeof(float));
  run_block(c, c.lv[lvl].blocks2[0], y,  b1, N, 2 * p, p, c.nbr[lvl]);
  run_block(c, c.lv[lvl].blocks2[1], b1, b2, N, p,     p, c.nbr[lvl]);
  return b2;
}

// --- parameter pytree walk (insertion order of reference dicts/lists) ---
static void parse_block(void* const* d_in, int& pi, BlkP& b, bool has_nin) {
  b.bn1.g = (const float*)d_in[pi++];
  b.bn1.b = (const float*)d_in[pi++];
  b.w1    = (const float*)d_in[pi++];
  b.bn2.g = (const float*)d_in[pi++];
  b.bn2.b = (const float*)d_in[pi++];
  b.w2    = (const float*)d_in[pi++];
  b.nin   = has_nin ? (const float*)d_in[pi++] : nullptr;
}

static void parse_unet(void* const* d_in, int& pi, Ctx& c, int lvl) {
  for (int i = 0; i < 2; ++i) parse_block(d_in, pi, c.lv[lvl].blocks[i], false);
  if (lvl >= 6) return;
  c.lv[lvl].bn_down.g = (const float*)d_in[pi++];
  c.lv[lvl].bn_down.b = (const float*)d_in[pi++];
  c.lv[lvl].down      = (const float*)d_in[pi++];
  parse_unet(d_in, pi, c, lvl + 1);
  c.lv[lvl].bn_up.g = (const float*)d_in[pi++];
  c.lv[lvl].bn_up.b = (const float*)d_in[pi++];
  c.lv[lvl].up      = (const float*)d_in[pi++];
  parse_block(d_in, pi, c.lv[lvl].blocks2[0], true);
  parse_block(d_in, pi, c.lv[lvl].blocks2[1], false);
}

extern "C" void kernel_launch(void* const* d_in, const int* in_sizes, int n_in,
                              void* d_out, int out_size, void* d_ws, size_t ws_size,
                              hipStream_t stream) {
  Ctx c;
  c.s = stream;
  c.ws = (char*)d_ws; c.off = 0; c.cap = ws_size;

  const float* feats = (const float*)d_in[0];
  const int N0 = in_sizes[0] / 3;
  for (int l = 0; l < 7; ++l) {
    c.Nl[l]  = in_sizes[1 + l] / 27;
    c.nbr[l] = (const int*)d_in[1 + l];
  }
  int pi = 8;
  for (int l = 0; l < 6; ++l) {
    for (int k = 0; k < 8; ++k) {
      c.didx[l][k] = (const int*)d_in[pi];
      c.dlen[l][k] = in_sizes[pi];
      pi++;
      c.dpar[l][k] = (const int*)d_in[pi++];
    }
  }
  c.stem = (const float*)d_in[pi++];
  parse_unet(d_in, pi, c, 0);
  c.bn_out.g = (const float*)d_in[pi++];
  c.bn_out.b = (const float*)d_in[pi++];
  c.lin_w  = (const float*)d_in[pi++];
  c.lin_b  = (const float*)d_in[pi++];
  c.lin1_w = (const float*)d_in[pi++];
  c.lin1_b = (const float*)d_in[pi++];
  c.cen1_w = (const float*)d_in[pi++];
  c.cen1_b = (const float*)d_in[pi++];
  c.cen2_w = (const float*)d_in[pi++];
  c.cen2_b = (const float*)d_in[pi++];

  // --- stem: feats [N0,3] (K padded to 32) -> x0 [N0,32] ---
  float* x0 = (float*)c.alloc((size_t)N0 * 32 * sizeof(float));
  {
    size_t mk = c.off;
    __bf16* fb = to_bf16_pad(c, feats, N0, 3, 32);
    __bf16* Ws = pack_w(c, c.stem, 27, /*Ksrc*/3, /*Kt*/1, 32, /*ntMajor*/1);
    subconv(c, fb, c.nbr[0], Ws, x0, N0, 32, 32);
    c.off = mk;
  }

  // --- U-Net ---
  float* xf = run_u(c, 0, x0);

  // --- heads: d_out = [ y(N0x20) | fv(N0x32) | off(N0x3) ] ---
  {
    size_t mk = c.off;
    float* outY   = (float*)d_out;
    float* outFV  = outY  + (size_t)N0 * 20;
    float* outOFF = outFV + (size_t)N0 * 32;

    __bf16* fvb = bnrelu_bf(c, xf, N0, 32, c.bn_out);

    __bf16* WL = pack_w(c, c.lin_w, 1, 32, 1, 20, 0);
    gemm(c, fvb, N0, nullptr, WL, c.lin_b, outY, nullptr, 32, 20, 20, 0);

    __bf16* W1 = pack_w(c, c.lin1_w, 1, 32, 1, 32, 0);
    gemm(c, fvb, N0, nullptr, W1, c.lin1_b, outFV, nullptr, 32, 32, 32, 0);

    __bf16* fvo = to_bf16_pad(c, outFV, N0, 32, 32);
    __bf16* Wc1 = pack_w(c, c.cen1_w, 1, 32, 1, 32, 0);
    float* h = (float*)c.alloc((size_t)N0 * 32 * sizeof(float));
    gemm(c, fvo, N0, nullptr, Wc1, c.cen1_b, h, nullptr, 32, 32, 32, /*relu*/2);

    __bf16* hb  = to_bf16_pad(c, h, N0, 32, 32);
    __bf16* Wc2 = pack_w(c, c.cen2_w, 1, 32, 1, 3, 0);
    gemm(c, hb, N0, nullptr, Wc2, c.cen2_b, outOFF, nullptr, 32, 3, 3, 0);
    c.off = mk;
  }
}
